// GNN_46557445488897
// MI455X (gfx1250) — matrix operand order repaired and checked by reading, not compiled, not run on hardware
//
#include <hip/hip_runtime.h>
#include <hip/hip_bf16.h>

#define L_DIM 256
#define H_DIM 512
#define G_NUM 512
#define NPG_NUM 32
#define EPG_NUM 64
#define N_NUM (G_NUM * NPG_NUM)   // 16384
#define E_NUM (G_NUM * EPG_NUM)   // 32768
#define N_STEPS 12

typedef __bf16 v16bf __attribute__((ext_vector_type(16)));
typedef float  v8f   __attribute__((ext_vector_type(8)));

// ---------------------------------------------------------------------------
// helpers
// ---------------------------------------------------------------------------
__device__ __forceinline__ unsigned bf16_rne(float f) {
  unsigned u = __float_as_uint(f);
  return (u + 0x7FFFu + ((u >> 16) & 1u)) >> 16;
}
__device__ __forceinline__ unsigned pack2(float a, float b) {
  return bf16_rne(a) | (bf16_rne(b) << 16);
}

union alignas(32) FragBF { unsigned u[8]; uint4 q[2]; v16bf v; };

// ---------------------------------------------------------------------------
// WMMA GEMM:  C[M x Nc] = relu(Abf16[M x K] @ Wbf16[K x Nc] + bias)
//
// A: row-major bf16 (== packed pairs, K/2 dwords per 16B-aligned row).
// W: fragment-major pack: Wf[((kb*(Nc/16)+nt)*32+lane)*8 + r]
//    so each lane's B fragment is 8 contiguous dwords (2 x b128 loads).
// Wave computes 32(M) x 64(N): 2 A frags x 4 B frags -> 8 WMMA / k-step.
// Block = 4 waves = 128 rows. K, Nc compile-time -> full unroll, imm offsets.
// Epilogue stores f32 (Cf) and/or bf16 (Cb) with bias+ReLU fused.
// ---------------------------------------------------------------------------
template <int K, int Nc>
__global__ __launch_bounds__(128) void k_gemm_bf16(
    const unsigned* __restrict__ A, const unsigned* __restrict__ B,
    const float* __restrict__ bias, float* __restrict__ Cf,
    unsigned short* __restrict__ Cb, int M) {
  constexpr int Kp = K / 2;                  // dwords per A row
  const int lane   = threadIdx.x & 31;
  const int wave   = threadIdx.x >> 5;
  const int lane16 = lane & 15;
  const int hi     = lane >> 4;              // 0: lanes 0-15, 1: lanes 16-31
  const int m0     = (blockIdx.y * 4 + wave) * 32;
  const int n0     = blockIdx.x * 64;
  if (m0 >= M) return;                        // wave-uniform

  const unsigned* Arow0 = A + (size_t)(m0 + lane16) * Kp;
  const unsigned* Arow1 = Arow0 + (size_t)16 * Kp;
  const unsigned* Bbase = B + ((size_t)(n0 >> 4) * 32 + lane) * 8;

  v8f acc[2][4] = {};
#pragma unroll
  for (int kb = 0; kb < K / 32; ++kb) {
    const int kp = kb * 16;
    FragBF a0, a1, b[4];
    a0.q[0] = *(const uint4*)(Arow0 + kp + hi * 4);
    a0.q[1] = *(const uint4*)(Arow0 + kp + 8 + hi * 4);
    a1.q[0] = *(const uint4*)(Arow1 + kp + hi * 4);
    a1.q[1] = *(const uint4*)(Arow1 + kp + 8 + hi * 4);
#pragma unroll
    for (int t = 0; t < 4; ++t) {
      const unsigned* Bp = Bbase + ((size_t)kb * (Nc / 16) + t) * 256;
      b[t].q[0] = *(const uint4*)(Bp);
      b[t].q[1] = *(const uint4*)(Bp + 4);
    }
#pragma unroll
    for (int t = 0; t < 4; ++t) {
      acc[0][t] = __builtin_amdgcn_wmma_f32_16x16x32_bf16(
          false, a0.v, false, b[t].v, (short)0, acc[0][t], false, false);
      acc[1][t] = __builtin_amdgcn_wmma_f32_16x16x32_bf16(
          false, a1.v, false, b[t].v, (short)0, acc[1][t], false, false);
    }
  }
  // C layout: lanes0-15 N=lane Vr->M=r ; lanes16-31 N=lane-16 Vr->M=8+r
#pragma unroll
  for (int i = 0; i < 2; ++i) {
#pragma unroll
    for (int t = 0; t < 4; ++t) {
      const int n = n0 + t * 16 + lane16;
      const float bv = bias[n];
#pragma unroll
      for (int r = 0; r < 8; ++r) {
        const int m = m0 + i * 16 + hi * 8 + r;
        float v = fmaxf(acc[i][t][r] + bv, 0.f);
        if (Cf) Cf[(size_t)m * Nc + n] = v;
        if (Cb) Cb[(size_t)m * Nc + n] = (unsigned short)bf16_rne(v);
      }
    }
  }
}

// ---------------------------------------------------------------------------
// weight pre-pack into fragment-major bf16 layout
// ---------------------------------------------------------------------------
__global__ void k_pack_weight_frag(const float* __restrict__ W,
                                   unsigned* __restrict__ Wf, int K, int Nc) {
  int idx = blockIdx.x * blockDim.x + threadIdx.x;
  int tot = (K >> 1) * Nc;
  if (idx >= tot) return;
  int r      = idx & 7;
  int lane   = (idx >> 3) & 31;
  int rest   = idx >> 8;                    // kb * (Nc/16) + nt
  int ntiles = Nc >> 4;
  int kb = rest / ntiles, nt = rest - kb * ntiles;
  int n  = nt * 16 + (lane & 15);
  int kp = kb * 16 + (r >> 2) * 8 + (lane >> 4) * 4 + (r & 3);
  Wf[idx] = pack2(W[(size_t)(2 * kp) * Nc + n], W[(size_t)(2 * kp + 1) * Nc + n]);
}

// ---------------------------------------------------------------------------
// small / data-movement kernels
// ---------------------------------------------------------------------------
__global__ void k_f32_to_bf16pack(const float* __restrict__ s,
                                  unsigned* __restrict__ d, int np) {
  int i = blockIdx.x * blockDim.x + threadIdx.x;
  if (i < np) d[i] = pack2(s[2 * i], s[2 * i + 1]);
}
__global__ void k_zero(float* __restrict__ a, int n) {
  int i = blockIdx.x * blockDim.x + threadIdx.x;
  if (i < n) a[i] = 0.f;
}
__global__ void k_add_out(const float* __restrict__ a, const float* __restrict__ b,
                          float* __restrict__ c, int n) {
  int i = blockIdx.x * blockDim.x + threadIdx.x;
  if (i < n) c[i] = a[i] + b[i];
}
__global__ void k_add_in(float* __restrict__ a, const float* __restrict__ b, int n) {
  int i = blockIdx.x * blockDim.x + threadIdx.x;
  if (i < n) a[i] += b[i];
}
__global__ void k_broadcast(const float* __restrict__ g0, float* __restrict__ u) {
  u[(size_t)blockIdx.x * L_DIM + threadIdx.x] = g0[threadIdx.x];
}

// one-hot concat @ W == gather-sum of W rows; emits bf16. blockDim = H.
__constant__ int c_atom_off[9] = {0, 119, 123, 135, 147, 157, 163, 169, 171};
__constant__ int c_bond_off[3] = {0, 5, 11};
__global__ void k_onehot_mlp1(const int* __restrict__ x, const float* __restrict__ W,
                              const float* __restrict__ b,
                              unsigned short* __restrict__ out,
                              int nf, int H, int isNode) {
  int n = blockIdx.x, j = threadIdx.x;
  float acc = b[j];
  for (int f = 0; f < nf; ++f) {
    int off = isNode ? c_atom_off[f] : c_bond_off[f];
    acc += W[(size_t)(off + x[n * nf + f]) * H + j];
  }
  out[(size_t)n * H + j] = (unsigned short)bf16_rne(fmaxf(acc, 0.f));
}

// pos_emb layer 1 (K=3) -> bf16.  blockDim = L.
__global__ void k_pos_emb1(const float* __restrict__ pos, const float* __restrict__ W,
                           const float* __restrict__ b,
                           unsigned short* __restrict__ out) {
  int n = blockIdx.x, j = threadIdx.x;
  float acc = b[j] + pos[n * 3 + 0] * W[j] + pos[n * 3 + 1] * W[L_DIM + j] +
              pos[n * 3 + 2] * W[2 * L_DIM + j];
  out[(size_t)n * L_DIM + j] = (unsigned short)bf16_rne(fmaxf(acc, 0.f));
}

// pairwise distance + dis_emb layer 1 (K=1) -> bf16.  blockDim = L.
__global__ void k_dis_emb1(const float* __restrict__ pos, const int* __restrict__ row,
                           const int* __restrict__ col, const float* __restrict__ W,
                           const float* __restrict__ b,
                           unsigned short* __restrict__ out) {
  int e = blockIdx.x, j = threadIdx.x;
  int r = row[e], c = col[e];
  float dx = pos[r * 3 + 0] - pos[c * 3 + 0];
  float dy = pos[r * 3 + 1] - pos[c * 3 + 1];
  float dz = pos[r * 3 + 2] - pos[c * 3 + 2];
  float d  = sqrtf(dx * dx + dy * dy + dz * dz);
  out[(size_t)e * L_DIM + j] = (unsigned short)bf16_rne(fmaxf(b[j] + d * W[j], 0.f));
}

// gather + bf16-pack concat builders (pair index p, 128 pairs per L segment)
__global__ void k_concat_edge(const float* __restrict__ extx, const float* __restrict__ exte,
                              const float* __restrict__ u, const int* __restrict__ row,
                              const int* __restrict__ col, const int* __restrict__ eb,
                              unsigned* __restrict__ A) {
  int e = blockIdx.x, p = threadIdx.x;      // p in [0,512)
  int seg = p >> 7, off = p & 127;
  const float* src;
  if (seg == 0)      src = extx + (size_t)row[e] * L_DIM;
  else if (seg == 1) src = extx + (size_t)col[e] * L_DIM;
  else if (seg == 2) src = exte + (size_t)e * L_DIM;
  else               src = u + (size_t)eb[e] * L_DIM;
  A[(size_t)e * 512 + p] = pack2(src[2 * off], src[2 * off + 1]);
}
__global__ void k_concat_node(const float* __restrict__ extx, const float* __restrict__ sent,
                              const float* __restrict__ recv, const float* __restrict__ u,
                              const int* __restrict__ nb, unsigned* __restrict__ A) {
  int n = blockIdx.x, p = threadIdx.x;      // p in [0,512)
  int seg = p >> 7, off = p & 127;
  const float* src;
  if (seg == 0)      src = extx + (size_t)n * L_DIM;
  else if (seg == 1) src = sent + (size_t)n * L_DIM;
  else if (seg == 2) src = recv + (size_t)n * L_DIM;
  else               src = u + (size_t)nb[n] * L_DIM;
  A[(size_t)n * 512 + p] = pack2(src[2 * off], src[2 * off + 1]);
}
__global__ void k_concat_glob(const float* __restrict__ u, const float* __restrict__ gn,
                              const float* __restrict__ ge, unsigned* __restrict__ A) {
  int g = blockIdx.x, p = threadIdx.x;      // p in [0,384)
  int seg = p >> 7, off = p & 127;
  const float* src = (seg == 0) ? u + (size_t)g * L_DIM
                   : (seg == 1) ? gn + (size_t)g * L_DIM
                                : ge + (size_t)g * L_DIM;
  A[(size_t)g * 384 + p] = pack2(src[2 * off], src[2 * off + 1]);
}

// scatter-add edges -> nodes (global_atomic_add_f32).  blockDim = L.
__global__ void k_scatter_add(const float* __restrict__ e1, const int* __restrict__ idx,
                              float* __restrict__ dst) {
  int e = blockIdx.x, j = threadIdx.x;
  atomicAdd(&dst[(size_t)idx[e] * L_DIM + j], e1[(size_t)e * L_DIM + j]);
}

// per-graph pooling over contiguous rows.  blockDim = L.
__global__ void k_segsum(const float* __restrict__ s, float* __restrict__ d, int per) {
  int g = blockIdx.x, j = threadIdx.x;
  float acc = 0.f;
  for (int i = 0; i < per; ++i) acc += s[((size_t)g * per + i) * L_DIM + j];
  d[(size_t)g * L_DIM + j] = acc;
}

// pos head layer 2: [L -> 3], no activation
__global__ void k_pos_out(const float* __restrict__ hp, const float* __restrict__ W,
                          const float* __restrict__ b, float* __restrict__ raw, int N) {
  int idx = blockIdx.x * blockDim.x + threadIdx.x;
  if (idx >= N * 3) return;
  int n = idx / 3, c = idx - 3 * n;
  float acc = b[c];
  const float* hrow = hp + (size_t)n * L_DIM;
  for (int k = 0; k < L_DIM; ++k) acc += hrow[k] * W[k * 3 + c];
  raw[idx] = acc;
}

// subtract per-graph centroid (32 nodes/graph).  blockDim = NPG.
__global__ void k_move2origin(const float* __restrict__ raw, float* __restrict__ pos) {
  __shared__ float sx[NPG_NUM], sy[NPG_NUM], sz[NPG_NUM], sm[3];
  int g = blockIdx.x, t = threadIdx.x;
  size_t base = ((size_t)g * NPG_NUM + t) * 3;
  float x = raw[base + 0], y = raw[base + 1], z = raw[base + 2];
  sx[t] = x; sy[t] = y; sz[t] = z;
  __syncthreads();
  if (t == 0) {
    float a = 0.f, b = 0.f, c = 0.f;
    for (int i = 0; i < NPG_NUM; ++i) { a += sx[i]; b += sy[i]; c += sz[i]; }
    sm[0] = a / NPG_NUM; sm[1] = b / NPG_NUM; sm[2] = c / NPG_NUM;
  }
  __syncthreads();
  pos[base + 0] = x - sm[0];
  pos[base + 1] = y - sm[1];
  pos[base + 2] = z - sm[2];
}

// ---------------------------------------------------------------------------
// host orchestration
// ---------------------------------------------------------------------------
extern "C" void kernel_launch(void* const* d_in, const int* in_sizes, int n_in,
                              void* d_out, int out_size, void* d_ws, size_t ws_size,
                              hipStream_t stream) {
  (void)in_sizes; (void)n_in; (void)out_size; (void)ws_size;
  const int N = N_NUM, E = E_NUM, G = G_NUM, L = L_DIM, H = H_DIM;

  const int*   x    = (const int*)d_in[0];
  const int*   row  = (const int*)d_in[1];
  const int*   col  = row + E;
  const int*   ea   = (const int*)d_in[2];
  const int*   nb   = (const int*)d_in[3];
  const int*   eb   = (const int*)d_in[4];
  const float* pos0 = (const float*)d_in[5];

  // param leaf indices (setup_inputs dict insertion order, (W,b) per layer)
  const int P_EN = 6, P_EE = 12, P_PE = 18, P_DE = 22, P_GI = 26, P_L0 = 27;

  // ---- bump allocator over d_ws ----
  char* wp = (char*)d_ws;
  auto alloc = [&](size_t bytes) -> void* {
    void* p = (void*)wp;
    wp += (bytes + 255) & ~(size_t)255;
    return p;
  };
  auto fbuf = [&](size_t n) { return (float*)alloc(n * 4); };
  auto ubuf = [&](size_t n) { return (unsigned*)alloc(n * 4); };
  auto hbuf = [&](size_t n) { return (unsigned short*)alloc(n * 2); };

  float *h    = fbuf((size_t)N * L), *e    = fbuf((size_t)E * L), *u   = fbuf((size_t)G * L);
  float *pos  = fbuf((size_t)N * 3), *praw = fbuf((size_t)N * 3);
  float *extx = fbuf((size_t)N * L), *exte = fbuf((size_t)E * L);
  float *e1b  = fbuf((size_t)E * L), *h1b  = fbuf((size_t)N * L), *u1b = fbuf((size_t)G * L);
  float *sent = fbuf((size_t)N * L), *recv = fbuf((size_t)N * L);
  float *gn   = fbuf((size_t)G * L), *ge   = fbuf((size_t)G * L);
  float          *tf  = fbuf((size_t)E * L);         // f32 scratch
  unsigned short *tbA = hbuf((size_t)E * H);         // bf16 scratch ping
  unsigned short *tbB = hbuf((size_t)E * H);         // bf16 scratch pong
  unsigned       *tA  = ubuf((size_t)E * (4 * L) / 2);// concat inputs (bf16 pairs)

  auto bias  = [&](int i) { return (const float*)d_in[i]; };
  auto packW = [&](int i, int K, int Nc) -> unsigned* {
    unsigned* dst = ubuf((size_t)(K / 2) * Nc);
    int tot = (K / 2) * Nc;
    k_pack_weight_frag<<<(tot + 255) / 256, 256, 0, stream>>>(bias(i), dst, K, Nc);
    return dst;
  };
  auto gemm = [&](const void* A, const unsigned* W, const float* b, float* Cf,
                  unsigned short* Cb, int M, int K, int Nc) {
    dim3 grid(Nc / 64, M / 128);
    const unsigned* Au = (const unsigned*)A;
    if (K == 1024 && Nc == 512)
      k_gemm_bf16<1024, 512><<<grid, 128, 0, stream>>>(Au, W, b, Cf, Cb, M);
    else if (K == 768 && Nc == 512)
      k_gemm_bf16<768, 512><<<grid, 128, 0, stream>>>(Au, W, b, Cf, Cb, M);
    else if (K == 512 && Nc == 512)
      k_gemm_bf16<512, 512><<<grid, 128, 0, stream>>>(Au, W, b, Cf, Cb, M);
    else if (K == 512 && Nc == 256)
      k_gemm_bf16<512, 256><<<grid, 128, 0, stream>>>(Au, W, b, Cf, Cb, M);
    else
      k_gemm_bf16<256, 256><<<grid, 128, 0, stream>>>(Au, W, b, Cf, Cb, M);
  };
  auto conv = [&](const float* s, void* d, int np) {
    k_f32_to_bf16pack<<<(np + 255) / 256, 256, 0, stream>>>(s, (unsigned*)d, np);
  };
  auto addout = [&](const float* a, const float* b2, float* c, int n) {
    k_add_out<<<(n + 255) / 256, 256, 0, stream>>>(a, b2, c, n);
  };
  auto addin = [&](float* a, const float* b2, int n) {
    k_add_in<<<(n + 255) / 256, 256, 0, stream>>>(a, b2, n);
  };
  auto zero = [&](float* a, int n) {
    k_zero<<<(n + 255) / 256, 256, 0, stream>>>(a, n);
  };

  // ---- pack all WMMA weights to fragment-major bf16 (idempotent per call) ----
  unsigned* w_en1 = packW(P_EN + 2, H, H);
  unsigned* w_en2 = packW(P_EN + 4, H, L);
  unsigned* w_ee1 = packW(P_EE + 2, H, H);
  unsigned* w_ee2 = packW(P_EE + 4, H, L);
  unsigned* w_pe1 = packW(P_PE + 2, L, L);
  unsigned* w_de1 = packW(P_DE + 2, L, L);
  unsigned *we0[N_STEPS], *we1[N_STEPS], *we2[N_STEPS];
  unsigned *wn0[N_STEPS], *wn1[N_STEPS], *wn2[N_STEPS];
  unsigned *wg0[N_STEPS], *wg1[N_STEPS], *wg2[N_STEPS], *wp0[N_STEPS];
  for (int l = 0; l < N_STEPS; ++l) {
    int b0 = P_L0 + 22 * l;
    we0[l] = packW(b0 + 0, 4 * L, H); we1[l] = packW(b0 + 2, H, H); we2[l] = packW(b0 + 4, H, L);
    wn0[l] = packW(b0 + 6, 4 * L, H); wn1[l] = packW(b0 + 8, H, H); wn2[l] = packW(b0 + 10, H, L);
    wg0[l] = packW(b0 + 12, 3 * L, H); wg1[l] = packW(b0 + 14, H, H); wg2[l] = packW(b0 + 16, H, L);
    wp0[l] = packW(b0 + 18, L, L);
  }

  // ---- encoders ----
  k_onehot_mlp1<<<N, H, 0, stream>>>(x, bias(P_EN), bias(P_EN + 1), tbA, 9, H, 1);
  gemm(tbA, w_en1, bias(P_EN + 3), nullptr, tbB, N, H, H);
  gemm(tbB, w_en2, bias(P_EN + 5), h, nullptr, N, H, L);

  k_onehot_mlp1<<<E, H, 0, stream>>>(ea, bias(P_EE), bias(P_EE + 1), tbA, 3, H, 0);
  gemm(tbA, w_ee1, bias(P_EE + 3), nullptr, tbB, E, H, H);
  gemm(tbB, w_ee2, bias(P_EE + 5), e, nullptr, E, H, L);

  k_broadcast<<<G, L, 0, stream>>>(bias(P_GI), u);
  (void)hipMemcpyAsync(pos, pos0, (size_t)N * 3 * 4, hipMemcpyDeviceToDevice, stream);

  // ---- message-passing steps ----
  for (int l = 0; l < N_STEPS; ++l) {
    int b0 = P_L0 + 22 * l;

    // ext_x = h + pos_emb(pos)
    k_pos_emb1<<<N, L, 0, stream>>>(pos, bias(P_PE), bias(P_PE + 1), tbA);
    gemm(tbA, w_pe1, bias(P_PE + 3), tf, nullptr, N, L, L);
    addout(h, tf, extx, N * L);

    // ext_e = e + dis_emb(|pos[row]-pos[col]|)
    k_dis_emb1<<<E, L, 0, stream>>>(pos, row, col, bias(P_DE), bias(P_DE + 1), tbA);
    gemm(tbA, w_de1, bias(P_DE + 3), tf, nullptr, E, L, L);
    addout(e, tf, exte, E * L);

    // edge model
    k_concat_edge<<<E, 512, 0, stream>>>(extx, exte, u, row, col, eb, tA);
    gemm(tA, we0[l], bias(b0 + 1), nullptr, tbA, E, 4 * L, H);
    gemm(tbA, we1[l], bias(b0 + 3), nullptr, tbB, E, H, H);
    gemm(tbB, we2[l], bias(b0 + 5), e1b, nullptr, E, H, L);

    // node model
    zero(sent, N * L);
    zero(recv, N * L);
    k_scatter_add<<<E, L, 0, stream>>>(e1b, row, sent);
    k_scatter_add<<<E, L, 0, stream>>>(e1b, col, recv);
    k_concat_node<<<N, 512, 0, stream>>>(extx, sent, recv, u, nb, tA);
    gemm(tA, wn0[l], bias(b0 + 7), nullptr, tbA, N, 4 * L, H);
    gemm(tbA, wn1[l], bias(b0 + 9), nullptr, tbB, N, H, H);
    gemm(tbB, wn2[l], bias(b0 + 11), h1b, nullptr, N, H, L);

    // global model
    k_segsum<<<G, L, 0, stream>>>(h1b, gn, NPG_NUM);
    k_segsum<<<G, L, 0, stream>>>(e1b, ge, EPG_NUM);
    k_concat_glob<<<G, 384, 0, stream>>>(u, gn, ge, tA);
    gemm(tA, wg0[l], bias(b0 + 13), nullptr, tbA, G, 3 * L, H);
    gemm(tbA, wg1[l], bias(b0 + 15), nullptr, tbB, G, H, H);
    gemm(tbB, wg2[l], bias(b0 + 17), u1b, nullptr, G, H, L);

    // residuals
    addin(h, h1b, N * L);
    addin(e, e1b, E * L);
    addin(u, u1b, G * L);

    // position head + move2origin
    conv(h, tbA, N * L / 2);
    gemm(tbA, wp0[l], bias(b0 + 19), tf, nullptr, N, L, L);
    k_pos_out<<<(N * 3 + 255) / 256, 256, 0, stream>>>(tf, bias(b0 + 20), bias(b0 + 21), praw, N);
    k_move2origin<<<G, NPG_NUM, 0, stream>>>(praw, pos);
  }

  (void)hipMemcpyAsync(d_out, pos, (size_t)N * 3 * 4, hipMemcpyDeviceToDevice, stream);
}
